// LSTMPredictor_63273458205300
// MI455X (gfx1250) — matrix operand order repaired and checked by reading, not compiled, not run on hardware
//
#include <hip/hip_runtime.h>
#include <hip/hip_bf16.h>

#define LAYERS 10
#define HDIM   128
#define ENCDIM 4096
#define BATCH  8192
#define TSTEPS 12
#define NMODES 25

typedef __attribute__((ext_vector_type(16))) _Float16 v16h;
typedef __attribute__((ext_vector_type(8)))  _Float16 v8h;
typedef __attribute__((ext_vector_type(8)))  float    v8f;
typedef __attribute__((ext_vector_type(4)))  float    v4f;

static __device__ __forceinline__ v8f wmma_f16(v16h a, v16h b, v8f c) {
  // 8 args: (neg_a, A, neg_b, B, c_mod, C, reuse_a, reuse_b)
  return __builtin_amdgcn_wmma_f32_16x16x32_f16(false, a, false, b, (short)0, c, false, false);
}

static __device__ __forceinline__ float sigmoidf_(float x) {
  return 1.0f / (1.0f + __expf(-x));
}

// ---------------------------------------------------------------------------
// Pack a row-major f32 weight W[N][K] into WMMA B-fragment order (f16):
//   dst[((kc*NT + nt)*32 + lane)*16 + e] = W[nt*16 + lane%16][kc*32 + (lane/16)*8 + (e%8) + (e/8)*16]
// so each lane's 16-half fragment is one contiguous 32-byte load.
// ---------------------------------------------------------------------------
__global__ void pack_b_kernel(const float* __restrict__ src, _Float16* __restrict__ dst,
                              int N, int K, int total) {
  int idx = blockIdx.x * blockDim.x + threadIdx.x;
  if (idx >= total) return;
  int e    = idx & 15;
  int lane = (idx >> 4) & 31;
  int rest = idx >> 9;
  int NT   = N >> 4;
  int nt   = rest % NT;
  int kc   = rest / NT;
  int n = nt * 16 + (lane & 15);
  int k = kc * 32 + ((lane >> 4) << 3) + (e & 7) + ((e >> 3) << 4);
  dst[idx] = (_Float16)src[(size_t)n * K + k];
}

__global__ void bias_sum_kernel(const float* __restrict__ a, const float* __restrict__ b,
                                float* __restrict__ dst, int total) {
  int i = blockIdx.x * blockDim.x + threadIdx.x;
  if (i < total) dst[i] = a[i] + b[i];
}

// ---------------------------------------------------------------------------
// fc_in: enc16[B][H] = relu(A[B][ENC] @ W^T + b), f16 output.
// 8 waves/block, each wave: 16 rows x 128 cols, K loop of 32.
// B fragments batch-loaded (one clause) before the 8-WMMA burst.
// ---------------------------------------------------------------------------
__global__ void fc_in_kernel(const float* __restrict__ A, const _Float16* __restrict__ Wp,
                             const float* __restrict__ bias, _Float16* __restrict__ enc16) {
  int wave   = threadIdx.x >> 5;
  int lane   = threadIdx.x & 31;
  int nlo    = lane & 15;
  int hiHalf = lane >> 4;
  int hi8    = hiHalf * 8;
  int rowBase = blockIdx.x * 128 + wave * 16;
  int arow = rowBase + nlo;

  v8f acc[8];
#pragma unroll
  for (int nt = 0; nt < 8; ++nt)
#pragma unroll
    for (int r = 0; r < 8; ++r) acc[nt][r] = 0.0f;

  const float* ap = A + (size_t)arow * ENCDIM;
  for (int kc = 0; kc < ENCDIM / 32; ++kc) {
    int k0 = kc * 32 + hi8;
    v4f f0 = *(const v4f*)(ap + k0);
    v4f f1 = *(const v4f*)(ap + k0 + 4);
    v4f f2 = *(const v4f*)(ap + k0 + 16);
    v4f f3 = *(const v4f*)(ap + k0 + 20);

    // batch the 8 B-fragment loads so they land in one clause
    const _Float16* wp = Wp + (size_t)kc * 8 * 512 + lane * 16;
    v16h b[8];
#pragma unroll
    for (int nt = 0; nt < 8; ++nt) b[nt] = *(const v16h*)(wp + nt * 512);

    v16h a;
#pragma unroll
    for (int e = 0; e < 4; ++e) {
      a[e]      = (_Float16)f0[e];
      a[4 + e]  = (_Float16)f1[e];
      a[8 + e]  = (_Float16)f2[e];
      a[12 + e] = (_Float16)f3[e];
    }
#pragma unroll
    for (int nt = 0; nt < 8; ++nt) acc[nt] = wmma_f16(a, b[nt], acc[nt]);
  }
  int mloc = hiHalf * 8;
#pragma unroll
  for (int nt = 0; nt < 8; ++nt) {
    float bv = bias[nt * 16 + nlo];
#pragma unroll
    for (int r = 0; r < 8; ++r) {
      float v = acc[nt][r] + bv;
      v = v > 0.0f ? v : 0.0f;
      enc16[(size_t)(rowBase + mloc + r) * HDIM + nt * 16 + nlo] = (_Float16)v;
    }
  }
}

// ---------------------------------------------------------------------------
// Full LSTM stack. 4 waves/block, each wave owns 16 batch rows end-to-end:
//  - per layer: block stages 256 KB of packed Wih/Whh in LDS (reused 12 steps)
//  - c kept in 64 VGPRs, h in private 4 KB LDS scratch (D-layout -> A-layout)
//  - layer activations ping-pong through L2-resident global buffers
// ---------------------------------------------------------------------------
__global__ void lstm_stack_kernel(const _Float16* __restrict__ enc16,
                                  const _Float16* __restrict__ wpack, // [L][2][4][32][32][16]
                                  const float* __restrict__ bsum,     // [L][512]
                                  _Float16* __restrict__ buf0,
                                  _Float16* __restrict__ buf1) {
  extern __shared__ _Float16 smem[];
  const int WPL  = 2 * 4 * 32 * 32 * 16;   // 131072 halves of weights per layer
  int wave   = threadIdx.x >> 5;
  int lane   = threadIdx.x & 31;
  int nlo    = lane & 15;
  int hiHalf = lane >> 4;
  int hi8    = hiHalf * 8;
  int mloc   = hiHalf * 8;
  int rowBase = blockIdx.x * 64 + wave * 16;
  _Float16* hbuf = smem + WPL + wave * (16 * HDIM);

  v8f c[8];

  for (int l = 0; l < LAYERS; ++l) {
    // ---- stage this layer's packed weights into LDS (256 KB) ----
    {
      const uint4* src = (const uint4*)(wpack + (size_t)l * WPL);
      uint4* dst = (uint4*)smem;
      for (int i = threadIdx.x; i < WPL / 8; i += blockDim.x) dst[i] = src[i];
    }
    __syncthreads();

    // ---- reset recurrent state ----
#pragma unroll
    for (int jt = 0; jt < 8; ++jt)
#pragma unroll
      for (int r = 0; r < 8; ++r) c[jt][r] = 0.0f;
    {
      v8h z;
#pragma unroll
      for (int e = 0; e < 8; ++e) z[e] = (_Float16)0.0f;
      for (int i = lane; i < (16 * HDIM) / 8; i += 32) *(v8h*)(hbuf + i * 8) = z;
    }

    const _Float16* xin = (l == 0) ? enc16 : ((l & 1) ? buf0 : buf1);
    _Float16* yout = (l & 1) ? buf1 : buf0;
    const float* bptr = bsum + l * 512;

    for (int t = 0; t < TSTEPS; ++t) {
      // ---- A-fragments for x (global) and h (LDS), 16x32 layout ----
      const _Float16* xrow = (l == 0)
          ? (xin + (size_t)(rowBase + nlo) * HDIM)                     // time-constant input
          : (xin + ((size_t)(rowBase + nlo) * TSTEPS + t) * HDIM);
      const _Float16* hrow = hbuf + nlo * HDIM;
      v16h ax[4], ah[4];
#pragma unroll
      for (int kc = 0; kc < 4; ++kc) {
        int k0 = kc * 32 + hi8;
        v8h xlo = *(const v8h*)(xrow + k0);
        v8h xhi = *(const v8h*)(xrow + k0 + 16);
        ax[kc] = __builtin_shufflevector(xlo, xhi, 0,1,2,3,4,5,6,7,8,9,10,11,12,13,14,15);
        v8h hlo = *(const v8h*)(hrow + k0);
        v8h hhi = *(const v8h*)(hrow + k0 + 16);
        ah[kc] = __builtin_shufflevector(hlo, hhi, 0,1,2,3,4,5,6,7,8,9,10,11,12,13,14,15);
      }

      // ---- gates: pair i/f/g/o tiles for the same 16 unit columns ----
#pragma unroll 1
      for (int jt = 0; jt < 8; ++jt) {
        v8f g[4];
#pragma unroll
        for (int gi = 0; gi < 4; ++gi)
#pragma unroll
          for (int r = 0; r < 8; ++r) g[gi][r] = 0.0f;

#pragma unroll
        for (int gi = 0; gi < 4; ++gi) {
          int nt = gi * 8 + jt;
          const _Float16* wb = smem + (size_t)nt * 512 + lane * 16;
          // batch the 8 LDS fragment loads, then the 8-deep WMMA burst
          v16h bw[4], bu[4];
#pragma unroll
          for (int kc = 0; kc < 4; ++kc) {
            bw[kc] = *(const v16h*)(wb + (size_t)kc * (32 * 512));        // Wih^T frag
            bu[kc] = *(const v16h*)(wb + (size_t)(4 + kc) * (32 * 512));  // Whh^T frag
          }
#pragma unroll
          for (int kc = 0; kc < 4; ++kc) {
            g[gi] = wmma_f16(ax[kc], bw[kc], g[gi]);
            g[gi] = wmma_f16(ah[kc], bu[kc], g[gi]);
          }
        }

        float bi = bptr[0 * 128 + jt * 16 + nlo];
        float bf_ = bptr[1 * 128 + jt * 16 + nlo];
        float bg = bptr[2 * 128 + jt * 16 + nlo];
        float bo = bptr[3 * 128 + jt * 16 + nlo];
        _Float16* hst = hbuf + (size_t)mloc * HDIM + jt * 16 + nlo;
        _Float16* yst = yout + ((size_t)(rowBase + mloc) * TSTEPS + t) * HDIM + jt * 16 + nlo;
#pragma unroll
        for (int r = 0; r < 8; ++r) {
          float iv = sigmoidf_(g[0][r] + bi);
          float fv = sigmoidf_(g[1][r] + bf_);
          float gv = tanhf(g[2][r] + bg);
          float ov = sigmoidf_(g[3][r] + bo);
          float cv = fv * c[jt][r] + iv * gv;
          c[jt][r] = cv;
          float hv = ov * tanhf(cv);
          hst[(size_t)r * HDIM] = (_Float16)hv;
          yst[(size_t)r * TSTEPS * HDIM] = (_Float16)hv;
        }
      }
    }
    __syncthreads();  // before overwriting LDS weights with next layer
  }
}

// ---------------------------------------------------------------------------
// fc_out + reshape + softmax. One thread per (b, t).
// ---------------------------------------------------------------------------
__global__ void fc_out_kernel(const _Float16* __restrict__ y, const float* __restrict__ Wf,
                              const float* __restrict__ bf, float* __restrict__ out) {
  __shared__ _Float16 wsm[3 * NMODES * HDIM];  // 75 x 128 f16
  for (int i = threadIdx.x; i < 3 * NMODES * HDIM; i += blockDim.x)
    wsm[i] = (_Float16)Wf[i];
  __syncthreads();

  int idx = blockIdx.x * blockDim.x + threadIdx.x;
  if (idx >= BATCH * TSTEPS) return;
  int b = idx / TSTEPS;
  int t = idx - b * TSTEPS;

  const _Float16* yrow = y + (size_t)idx * HDIM;
  v8h yv[16];
#pragma unroll
  for (int i = 0; i < 16; ++i) yv[i] = *(const v8h*)(yrow + i * 8);

  const size_t PRED = (size_t)BATCH * NMODES * TSTEPS * 2;
  float pv[25];
  for (int j = 0; j < 75; ++j) {
    float acc = bf[j];
    const _Float16* wr = wsm + j * HDIM;
#pragma unroll 4
    for (int i = 0; i < 16; ++i) {
      v8h w = *(const v8h*)(wr + i * 8);
#pragma unroll
      for (int e = 0; e < 8; ++e) acc += (float)yv[i][e] * (float)w[e];
    }
    if (j < 50) {
      int m = j >> 1, d = j & 1;
      out[(((size_t)b * NMODES + m) * TSTEPS + t) * 2 + d] = acc;
    } else {
      pv[j - 50] = acc;
    }
  }
  if (t == TSTEPS - 1) {
    float mx = pv[0];
#pragma unroll
    for (int m = 1; m < 25; ++m) mx = fmaxf(mx, pv[m]);
    float s = 0.0f;
#pragma unroll
    for (int m = 0; m < 25; ++m) { pv[m] = __expf(pv[m] - mx); s += pv[m]; }
    float inv = 1.0f / s;
#pragma unroll
    for (int m = 0; m < 25; ++m) out[PRED + (size_t)b * NMODES + m] = pv[m] * inv;
  }
}

// ---------------------------------------------------------------------------
extern "C" void kernel_launch(void* const* d_in, const int* in_sizes, int n_in,
                              void* d_out, int out_size, void* d_ws, size_t ws_size,
                              hipStream_t stream) {
  const float* backbone = (const float*)d_in[0];
  const float* fc_in_W  = (const float*)d_in[1];
  const float* fc_in_b  = (const float*)d_in[2];
  const float* Wih      = (const float*)d_in[3];
  const float* Whh      = (const float*)d_in[4];
  const float* bih      = (const float*)d_in[5];
  const float* bhh      = (const float*)d_in[6];
  const float* fc_out_W = (const float*)d_in[7];
  const float* fc_out_b = (const float*)d_in[8];
  float* out = (float*)d_out;

  char* ws = (char*)d_ws;
  size_t off = 0;
  auto alloc = [&](size_t bytes) {
    char* p = ws + off;
    off += (bytes + 255) & ~(size_t)255;
    return p;
  };
  _Float16* enc16   = (_Float16*)alloc((size_t)BATCH * HDIM * 2);
  _Float16* wpackIn = (_Float16*)alloc((size_t)ENCDIM * HDIM * 2);
  _Float16* wpackL  = (_Float16*)alloc((size_t)LAYERS * 131072 * 2);
  float*    bsum    = (float*)   alloc((size_t)LAYERS * 512 * 4);
  _Float16* buf0    = (_Float16*)alloc((size_t)BATCH * TSTEPS * HDIM * 2);
  _Float16* buf1    = (_Float16*)alloc((size_t)BATCH * TSTEPS * HDIM * 2);

  // weight prepacking into WMMA B-fragment order
  pack_b_kernel<<<(ENCDIM * HDIM) / 256, 256, 0, stream>>>(fc_in_W, wpackIn, HDIM, ENCDIM,
                                                           ENCDIM * HDIM);
  for (int l = 0; l < LAYERS; ++l) {
    pack_b_kernel<<<65536 / 256, 256, 0, stream>>>(Wih + (size_t)l * 65536,
                                                   wpackL + (size_t)l * 131072, 512, 128, 65536);
    pack_b_kernel<<<65536 / 256, 256, 0, stream>>>(Whh + (size_t)l * 65536,
                                                   wpackL + (size_t)l * 131072 + 65536, 512, 128,
                                                   65536);
  }
  bias_sum_kernel<<<(LAYERS * 512 + 255) / 256, 256, 0, stream>>>(bih, bhh, bsum, LAYERS * 512);

  // memory-bound encoder FC (streams 134 MB once)
  fc_in_kernel<<<BATCH / 128, 256, 0, stream>>>(backbone, wpackIn, fc_in_b, enc16);

  // compute-bound LSTM stack, weights LDS-resident per layer
  size_t shmem = (size_t)(131072 + 4 * 16 * HDIM) * 2;  // 272 KB
  lstm_stack_kernel<<<BATCH / 64, 128, shmem, stream>>>(enc16, wpackL, bsum, buf0, buf1);

  // layer 9 (odd) wrote buf1
  fc_out_kernel<<<(BATCH * TSTEPS) / 256, 256, 0, stream>>>(buf1, fc_out_W, fc_out_b, out);
}
